// AS_Mamba_Block_14190571946063
// MI455X (gfx1250) — compile-verified, matrix operand order checked by reading, not run on hardware
//
#include <hip/hip_runtime.h>

typedef __attribute__((ext_vector_type(16))) __bf16 v16bf;
typedef __attribute__((ext_vector_type(8)))  float  v8f;

#define HH 28
#define WW 28
#define CC 128
#define GG 64
#define PP (HH*WW)
#define NDEPTH 2

// ---------- bf16 helpers (manual RNE conversion; storage = ushort) ----------
__device__ __forceinline__ unsigned short f2bf(float f) {
  union { float f; unsigned u; } t; t.f = f;
  unsigned u = t.u;
  unsigned r = (u + 0x7FFFu + ((u >> 16) & 1u)) >> 16;
  return (unsigned short)r;
}
__device__ __forceinline__ float bf2f(unsigned short s) {
  union { unsigned u; float f; } t; t.u = ((unsigned)s) << 16;
  return t.f;
}
__device__ __forceinline__ float siluf(float x) { return x / (1.f + __expf(-x)); }
__device__ __forceinline__ float softplusf(float x) { return (x > 20.f) ? x : log1pf(__expf(x)); }

// ---------- WMMA fragment load (16-bit A layout, row-major LDS src) ----------
// lane l: m = l&15, k-groups: VGPR0-3 <- k0 + (l>>4)*8 + 0..7, VGPR4-7 <- +16.
__device__ __forceinline__ v16bf load_frag(const unsigned short* base, int row0,
                                           int stride, int k0, int lane) {
  int m  = lane & 15;
  int kg = (lane >> 4) << 3;
  const unsigned short* r = base + (row0 + m) * stride + k0 + kg;
  union { v16bf v; uint4 u[2]; } t;
  t.u[0] = *(const uint4*)(r);
  t.u[1] = *(const uint4*)(r + 16);
  return t.v;
}

// ---------- tiled LDS GEMM: OUT(LPxN) (+)= epilogue(A(LPxK) @ B(KxN)) ----------
// BT is B transposed in LDS: [N][K] row-major. MODE: 0 store, 1 silu-store,
// 2 softplus(acc+bias[n])-store, 3 add-to-out, 4 silu-add-to-out.
// Each wave computes TWO adjacent N-tiles per job (shared A fragment feeds two
// independent WMMA chains, hiding the bf16 WMMA->WMMA hazard slots), and the
// K loop is unrolled x2 with distinct fragment variables so next-step
// ds_load_b128s don't WAR against in-flight WMMA source registers.
template<int MODE>
__device__ void gemm_lds(const unsigned short* A, int lda, const unsigned short* BT,
                         int LP, int N, int K, unsigned short* O, int ldo,
                         const float* colconst, int tid) {
  const int lane  = tid & 31;
  const int wv    = tid >> 5;
  const int ntile = N >> 4;
  const int npair = ntile >> 1;            // N is 128 or 64 -> even #tiles
  const int tiles = (LP >> 4) * npair;
  for (int job = wv; job < tiles; job += 8) {
    int mt = job / npair, np = job - mt * npair;
    int nt0 = np * 2, nt1 = np * 2 + 1;
    v8f acc0 = {}, acc1 = {};
    for (int k0 = 0; k0 < K; k0 += 64) {   // K is 64 or 128: exact
      v16bf a0  = load_frag(A,  mt * 16,  lda, k0,      lane);
      v16bf b00 = load_frag(BT, nt0 * 16, K,   k0,      lane);
      v16bf b01 = load_frag(BT, nt1 * 16, K,   k0,      lane);
      v16bf a1  = load_frag(A,  mt * 16,  lda, k0 + 32, lane);
      v16bf b10 = load_frag(BT, nt0 * 16, K,   k0 + 32, lane);
      v16bf b11 = load_frag(BT, nt1 * 16, K,   k0 + 32, lane);
      acc0 = __builtin_amdgcn_wmma_f32_16x16x32_bf16(false, a0, false, b00,
                                                     (short)0, acc0, false, false);
      acc1 = __builtin_amdgcn_wmma_f32_16x16x32_bf16(false, a0, false, b01,
                                                     (short)0, acc1, false, false);
      acc0 = __builtin_amdgcn_wmma_f32_16x16x32_bf16(false, a1, false, b10,
                                                     (short)0, acc0, false, false);
      acc1 = __builtin_amdgcn_wmma_f32_16x16x32_bf16(false, a1, false, b11,
                                                     (short)0, acc1, false, false);
    }
    int mbase = mt * 16 + ((lane >> 4) << 3);
    #pragma unroll
    for (int t = 0; t < 2; ++t) {
      int n = (t ? nt1 : nt0) * 16 + (lane & 15);
      float cc = (MODE == 2) ? colconst[n] : 0.f;
      #pragma unroll
      for (int v = 0; v < 8; ++v) {
        int r = mbase + v;
        float x = t ? acc1[v] : acc0[v];
        if (MODE == 0)      O[r * ldo + n] = f2bf(x);
        else if (MODE == 1) O[r * ldo + n] = f2bf(siluf(x));
        else if (MODE == 2) O[r * ldo + n] = f2bf(softplusf(x + cc));
        else if (MODE == 3) O[r * ldo + n] = f2bf(bf2f(O[r * ldo + n]) + x);
        else                O[r * ldo + n] = f2bf(bf2f(O[r * ldo + n]) + siluf(x));
      }
    }
  }
}

// stage global f32 weight [K][N] into LDS bf16 transposed [N][K]
__device__ void stage_wt(const float* Wsrc, int K, int N, unsigned short* WT, int tid) {
  for (int e = tid; e < K * N; e += 256) {
    int n = e / K, k = e - n * K;
    WT[e] = f2bf(Wsrc[k * N + n]);
  }
}

// =====================================================================
// Fused per-(pixel, span) kernel: gather -> 2x dual-input SSM block -> scatter
// =====================================================================
template<int SPAN>
__global__ __launch_bounds__(256) void span_block_kernel(
    const float* __restrict__ fm, const float* __restrict__ fg,
    const float* __restrict__ flow, const float* __restrict__ sx,
    const float* __restrict__ sy,
    const float* __restrict__ W_in, const float* __restrict__ W_z,
    const float* __restrict__ W_dt, const float* __restrict__ b_dt,
    const float* __restrict__ a_log, const float* __restrict__ W_out,
    const float* __restrict__ W_yg, int si,
    float* __restrict__ accM, float* __restrict__ accG, float* __restrict__ wsum) {
  constexpr int L  = SPAN * SPAN;
  constexpr int LP = (L + 15) & ~15;
  constexpr int HS = SPAN / 2;

  const int p  = blockIdx.x;
  const int yy = p / WW, xx = p - yy * WW;
  const int tid = threadIdx.x;

  // span-group mask (groups {15,9,5} are disjoint in [3,17))
  float avg = 0.5f * (sx[p] + sy[p]);
  if (fabsf(avg - (float)SPAN) >= 1.5f) return;

  float cy = fminf(fmaxf((float)yy + flow[p * 4 + 1], 0.f), (float)(HH - 1));
  float cx = fminf(fmaxf((float)xx + flow[p * 4 + 0], 0.f), (float)(WW - 1));

  extern __shared__ unsigned char smem_raw[];
  unsigned short* Mb = (unsigned short*)smem_raw;   // [LP][CC] bf16
  unsigned short* Gb = Mb + LP * CC;                // [LP][GG]
  unsigned short* Ub = Gb + LP * GG;                // [LP][CC]  u / y
  unsigned short* Db = Ub + LP * CC;                // [LP][CC]  staging / delta
  unsigned short* Zb = Db + LP * CC;                // [LP][CC]  z
  unsigned short* WT = Zb + LP * CC;                // [CC][CC]  weight tile
  int*   tapi = (int*)(WT + CC * CC);               // [LP*4]
  float* tapw = (float*)(tapi + LP * 4);            // [LP*4]
  float* rmsb = tapw + LP * 4;                      // [LP]

  // ---------------- Phase 1: bilinear tap precompute ----------------
  for (int l = tid; l < LP; l += 256) {
    if (l < L) {
      int dyl = l / SPAN - HS, dxl = l % SPAN - HS;
      float ys = cy + (float)dyl, xs = cx + (float)dxl;
      float y0 = floorf(ys), x0 = floorf(xs);
      float wy = ys - y0, wx = xs - x0;
      #pragma unroll
      for (int t = 0; t < 4; ++t) {
        float yi = y0 + (float)(t >> 1);
        float xi = x0 + (float)(t & 1);
        bool  ok = (yi >= 0.f) && (yi <= (float)(HH - 1)) &&
                   (xi >= 0.f) && (xi <= (float)(WW - 1));
        float w = ((t >> 1) ? wy : 1.f - wy) * ((t & 1) ? wx : 1.f - wx) *
                  (ok ? 1.f : 0.f);
        int iy = (int)fminf(fmaxf(yi, 0.f), (float)(HH - 1));
        int ix = (int)fminf(fmaxf(xi, 0.f), (float)(WW - 1));
        tapi[l * 4 + t] = iy * WW + ix;
        tapw[l * 4 + t] = w;
      }
    } else {
      #pragma unroll
      for (int t = 0; t < 4; ++t) { tapi[l * 4 + t] = 0; tapw[l * 4 + t] = 0.f; }
    }
  }
  __syncthreads();

  // gather windows (bf16)
  for (int e = tid; e < LP * CC; e += 256) {
    int l = e >> 7, c = e & (CC - 1);
    const float* f = fm + c * PP;
    float v = tapw[l*4+0]*f[tapi[l*4+0]] + tapw[l*4+1]*f[tapi[l*4+1]] +
              tapw[l*4+2]*f[tapi[l*4+2]] + tapw[l*4+3]*f[tapi[l*4+3]];
    Mb[e] = f2bf(v);
  }
  for (int e = tid; e < LP * GG; e += 256) {
    int l = e >> 6, c = e & (GG - 1);
    const float* f = fg + c * PP;
    float v = tapw[l*4+0]*f[tapi[l*4+0]] + tapw[l*4+1]*f[tapi[l*4+1]] +
              tapw[l*4+2]*f[tapi[l*4+2]] + tapw[l*4+3]*f[tapi[l*4+3]];
    Gb[e] = f2bf(v);
  }
  __syncthreads();

  // ---------------- Phase 2: DEPTH dual-input blocks ----------------
  for (int d = 0; d < NDEPTH; ++d) {
    const int   wofs   = (si * NDEPTH + d);
    const float* Wz_s  = W_z   + wofs * GG * CC;
    const float* Wi_s  = W_in  + wofs * CC * CC;
    const float* Wdt_s = W_dt  + wofs * CC * CC;
    const float* bdt_s = b_dt  + wofs * CC;
    const float* alg_s = a_log + wofs * CC;
    const float* Wo_s  = W_out + wofs * CC * CC;
    const float* Wyg_s = W_yg  + wofs * CC * GG;

    // z = silu(rmsnorm(g) @ Wz)
    stage_wt(Wz_s, GG, CC, WT, tid);
    for (int l = tid; l < LP; l += 256) {
      float s = 0.f;
      for (int c = 0; c < GG; ++c) { float x = bf2f(Gb[l * GG + c]); s += x * x; }
      rmsb[l] = rsqrtf(s * (1.f / GG) + 1e-5f);
    }
    __syncthreads();
    for (int e = tid; e < LP * GG; e += 256)
      Db[e] = f2bf(bf2f(Gb[e]) * rmsb[e >> 6]);
    __syncthreads();
    gemm_lds<1>(Db, GG, WT, LP, CC, GG, Zb, CC, nullptr, tid);
    __syncthreads();

    // u = rmsnorm(m) @ Wi
    stage_wt(Wi_s, CC, CC, WT, tid);
    for (int l = tid; l < LP; l += 256) {
      float s = 0.f;
      for (int c = 0; c < CC; ++c) { float x = bf2f(Mb[l * CC + c]); s += x * x; }
      rmsb[l] = rsqrtf(s * (1.f / CC) + 1e-5f);
    }
    __syncthreads();
    for (int e = tid; e < LP * CC; e += 256)
      Db[e] = f2bf(bf2f(Mb[e]) * rmsb[e >> 7]);
    __syncthreads();
    gemm_lds<0>(Db, CC, WT, LP, CC, CC, Ub, CC, nullptr, tid);
    __syncthreads();

    // delta = softplus(u @ Wdt + bdt)
    stage_wt(Wdt_s, CC, CC, WT, tid);
    __syncthreads();
    gemm_lds<2>(Ub, CC, WT, LP, CC, CC, Db, CC, bdt_s, tid);
    __syncthreads();

    // diagonal selective scan over L, then y = h*z (in place over u)
    if (tid < CC) {
      int c = tid;
      float na = -__expf(alg_s[c]);
      float h = 0.f;
      for (int l = 0; l < LP; ++l) {
        float delta = bf2f(Db[l * CC + c]);
        float u     = bf2f(Ub[l * CC + c]);
        float dec   = __expf(delta * na);
        h = dec * h + delta * u;
        Ub[l * CC + c] = f2bf(h * bf2f(Zb[l * CC + c]));
      }
    }
    __syncthreads();

    // m += y @ Wo
    stage_wt(Wo_s, CC, CC, WT, tid);
    __syncthreads();
    gemm_lds<3>(Ub, CC, WT, LP, CC, CC, Mb, CC, nullptr, tid);
    __syncthreads();

    // g += silu(y @ Wyg)
    stage_wt(Wyg_s, CC, GG, WT, tid);
    __syncthreads();
    gemm_lds<4>(Ub, CC, WT, LP, GG, CC, Gb, GG, nullptr, tid);
    __syncthreads();
  }

  // ---------------- Phase 3: weighted scatter-accumulate ----------------
  for (int l = tid; l < LP; l += 256) {
    float w = 0.f; int tg = 0;
    if (l < L) {
      int dyl = l / SPAN - HS, dxl = l % SPAN - HS;
      int ty = yy + dyl, tx = xx + dxl;
      if (ty >= 0 && ty < HH && tx >= 0 && tx < WW) {
        float s = 0.f;
        for (int j = 0; j < L; ++j) {
          int jy = j / SPAN - HS, jx = j % SPAN - HS;
          s += __expf(-sqrtf((float)(jy * jy + jx * jx)) * (2.f / SPAN));
        }
        w = __expf(-sqrtf((float)(dyl * dyl + dxl * dxl)) * (2.f / SPAN)) / s;
        tg = ty * WW + tx;
        atomicAdd(&wsum[tg], w);
      }
    }
    tapw[l] = w; tapi[l] = tg;
  }
  __syncthreads();
  for (int e = tid; e < L * CC; e += 256) {
    int l = e >> 7, c = e & (CC - 1);
    float w = tapw[l];
    if (w > 0.f) atomicAdd(&accM[tapi[l] * CC + c], bf2f(Mb[l * CC + c]) * w);
  }
  for (int e = tid; e < L * GG; e += 256) {
    int l = e >> 6, c = e & (GG - 1);
    float w = tapw[l];
    if (w > 0.f) atomicAdd(&accG[tapi[l] * GG + c], bf2f(Gb[l * GG + c]) * w);
  }
}

// =====================================================================
__global__ void zero_kernel(float* p, int n) {
  for (int i = blockIdx.x * blockDim.x + threadIdx.x; i < n; i += gridDim.x * blockDim.x)
    p[i] = 0.f;
}

__global__ void finalize_kernel(const float* __restrict__ acc, float* __restrict__ out) {
  const int NM = PP * CC, NG = PP * GG;
  const float* ws = acc + 2 * NM + 2 * NG;
  int i = blockIdx.x * blockDim.x + threadIdx.x;
  if (i < 2 * NM) {
    int s = i / NM, r = i - s * NM;
    int pp = r >> 7, c = r & (CC - 1);
    float w = fmaxf(ws[s * PP + pp], 1e-6f);
    out[s * NM + c * PP + pp] = acc[s * NM + r] / w;
  } else if (i < 2 * NM + 2 * NG) {
    int j = i - 2 * NM;
    int s = j / NG, r = j - s * NG;
    int pp = r >> 6, c = r & (GG - 1);
    float w = fmaxf(ws[s * PP + pp], 1e-6f);
    out[2 * NM + s * NG + c * PP + pp] = acc[2 * NM + j] / w;
  }
}

static inline size_t smem_for(int span) {
  int L = span * span, LP = (L + 15) & ~15;
  return (size_t)((LP * CC * 4 + LP * GG + CC * CC) * 2 + LP * 16 * 2 + LP * 4);
}

extern "C" void kernel_launch(void* const* d_in, const int* in_sizes, int n_in,
                              void* d_out, int out_size, void* d_ws, size_t ws_size,
                              hipStream_t stream) {
  const float* fm[2]  = {(const float*)d_in[0], (const float*)d_in[1]};
  const float* fg[2]  = {(const float*)d_in[2], (const float*)d_in[3]};
  const float* fl[2]  = {(const float*)d_in[4], (const float*)d_in[5]};
  const float* sxs[2] = {(const float*)d_in[6], (const float*)d_in[8]};
  const float* sys[2] = {(const float*)d_in[7], (const float*)d_in[9]};
  const float* Wi  = (const float*)d_in[10];
  const float* Wz  = (const float*)d_in[11];
  const float* Wdt = (const float*)d_in[12];
  const float* bdt = (const float*)d_in[13];
  const float* alg = (const float*)d_in[14];
  const float* Wo  = (const float*)d_in[15];
  const float* Wyg = (const float*)d_in[16];

  float* ws = (float*)d_ws;
  const int NM = PP * CC, NG = PP * GG;
  float* accM  = ws;
  float* accG  = ws + 2 * NM;
  float* wsumb = ws + 2 * NM + 2 * NG;
  const int nacc = 2 * NM + 2 * NG + 2 * PP;

  (void)hipFuncSetAttribute((const void*)span_block_kernel<15>,
                            hipFuncAttributeMaxDynamicSharedMemorySize, (int)smem_for(15));
  (void)hipFuncSetAttribute((const void*)span_block_kernel<9>,
                            hipFuncAttributeMaxDynamicSharedMemorySize, (int)smem_for(9));
  (void)hipFuncSetAttribute((const void*)span_block_kernel<5>,
                            hipFuncAttributeMaxDynamicSharedMemorySize, (int)smem_for(5));

  zero_kernel<<<dim3((nacc + 255) / 256), dim3(256), 0, stream>>>(ws, nacc);

  for (int side = 0; side < 2; ++side) {
    float* aM = accM + side * NM;
    float* aG = accG + side * NG;
    float* aW = wsumb + side * PP;
    span_block_kernel<15><<<dim3(PP), dim3(256), smem_for(15), stream>>>(
        fm[side], fg[side], fl[side], sxs[side], sys[side],
        Wi, Wz, Wdt, bdt, alg, Wo, Wyg, 0, aM, aG, aW);
    span_block_kernel<9><<<dim3(PP), dim3(256), smem_for(9), stream>>>(
        fm[side], fg[side], fl[side], sxs[side], sys[side],
        Wi, Wz, Wdt, bdt, alg, Wo, Wyg, 1, aM, aG, aW);
    span_block_kernel<5><<<dim3(PP), dim3(256), smem_for(5), stream>>>(
        fm[side], fg[side], fl[side], sxs[side], sys[side],
        Wi, Wz, Wdt, bdt, alg, Wo, Wyg, 2, aM, aG, aW);
  }

  const int nout = 2 * NM + 2 * NG;
  finalize_kernel<<<dim3((nout + 255) / 256), dim3(256), 0, stream>>>(ws, (float*)d_out);
}
